// ExpertParallelBehindBlock_71571335020920
// MI455X (gfx1250) — matrix operand order repaired and checked by reading, not compiled, not run on hardware
//
#include <hip/hip_runtime.h>
#include <hip/hip_bf16.h>
#include <stdint.h>

// ---------------------------------------------------------------------------
// MoE expert block for MI455X (gfx1250): bf16 WMMA, fp32 accumulate.
//   Kernel 1: per-expert GEMM + bias -> bf16 intermediate, stored TRANSPOSED
//             EoutT[D_OUT][E*CAP] (makes k1 epilogue contiguous b128 stores
//             and k2's B tile an async-copy-friendly contiguous load).
//   Kernel 2: combine GEMM  Out[S,N] = CW[S,K] @ EoutT^T  -> fp32 output.
// Tiling: block 128x128, BK=64, 256 threads (8 waves, 2x4), wave tile 64x32,
//         double-buffered LDS (padded rows), 16 WMMA per staging round.
//         Kernel 2 B tiles staged via GLOBAL_LOAD_ASYNC_TO_LDS (ASYNCcnt).
// ---------------------------------------------------------------------------

typedef __attribute__((ext_vector_type(16))) __bf16 v16bf;
typedef __attribute__((ext_vector_type(8)))  __bf16 v8bf;
typedef __attribute__((ext_vector_type(4)))  __bf16 v4bf;
typedef __attribute__((ext_vector_type(8)))  float  v8f;
typedef __attribute__((ext_vector_type(4)))  float  v4f;
typedef __attribute__((ext_vector_type(4)))  int    v4i;

#define BM 128
#define BN 128
#define BK 64
#define BKP 72   // padded LDS row stride in bf16 elems (144 B, 16B-aligned)
#define WM 64    // wave tile M (2 waves along M)
#define WN 32    // wave tile N (4 waves along N)
#define MT 4     // 16x16 tiles per wave in M
#define NT 2     // 16x16 tiles per wave in N

#if defined(__has_builtin)
#  if __has_builtin(__builtin_amdgcn_global_load_async_to_lds_b128)
#    define HAVE_ASYNC_LDS 1
#  endif
#  if __has_builtin(__builtin_amdgcn_s_wait_asynccnt)
#    define HAVE_WAIT_ASYNC 1
#  endif
#endif

__device__ __forceinline__ void wait_async_all() {
#if defined(HAVE_ASYNC_LDS)
#  if defined(HAVE_WAIT_ASYNC)
  __builtin_amdgcn_s_wait_asynccnt(0);
#  else
  asm volatile("s_wait_asynccnt 0x0" ::: "memory");
#  endif
#endif
}

#if defined(HAVE_ASYNC_LDS)
// Probe-discovered signature: param0 is addrspace(1) pointer to 16B int
// vector, param1 the LDS destination, then imm offset and cpol.
__device__ __forceinline__ void async_copy_b128(const void* g, void* l) {
  void* gv = const_cast<void*>(g);
  __builtin_amdgcn_global_load_async_to_lds_b128(
      (__attribute__((address_space(1))) v4i*)gv,
      (__attribute__((address_space(3))) v4i*)l, 0, 0);
}
#endif

union ABFrag { v16bf v; v8bf h[2]; };

// 16 WMMAs on one staged BK=64 slab.
// A LDS: row-major [m][BKP].  Lane (half=lane>>4, l16) A-frag per the ISA
//   16-bit A 16x32 layout: K = {kk+half*8..+7} and {kk+16+half*8..+7}.
// B LDS: transposed [n][BKP]. Lane B-frag in the same K order as A:
//   K = {kk+half*8..+7} and {kk+16+half*8..+7} of column n.
__device__ __forceinline__ void mma_slab(const __bf16* As_, const __bf16* Bs_,
                                         v8f acc[MT][NT], int wm, int wn,
                                         int half, int l16) {
#pragma unroll
  for (int kk = 0; kk < BK; kk += 32) {
    v16bf af[MT];
#pragma unroll
    for (int mt = 0; mt < MT; ++mt) {
      const int m = wm * WM + mt * 16 + l16;
      ABFrag f;
      f.h[0] = *(const v8bf*)(As_ + m * BKP + kk + half * 8);
      f.h[1] = *(const v8bf*)(As_ + m * BKP + kk + 16 + half * 8);
      af[mt] = f.v;
    }
#pragma unroll
    for (int nt = 0; nt < NT; ++nt) {
      const int n = wn * WN + nt * 16 + l16;
      ABFrag g;
      g.h[0] = *(const v8bf*)(Bs_ + n * BKP + kk + half * 8);
      g.h[1] = *(const v8bf*)(Bs_ + n * BKP + kk + 16 + half * 8);
#pragma unroll
      for (int mt = 0; mt < MT; ++mt)
        acc[mt][nt] = __builtin_amdgcn_wmma_f32_16x16x32_bf16(
            false, af[mt], false, g.v, (short)0, acc[mt][nt], false, false);
    }
  }
}

// ---------------------------------------------------------------------------
// Kernel 1: EoutT[n, e*CAP+c] = X[e,c,:] @ W[e,:,n] + bias[e,n]   (bf16)
// ---------------------------------------------------------------------------
__global__ __launch_bounds__(256) void expert_gemm_bias_kernel(
    const float* __restrict__ X,       // [E*CAP, D_IN]
    const float* __restrict__ W,       // [E, D_IN, D_OUT]
    const float* __restrict__ Bias,    // [E, D_OUT]
    __bf16* __restrict__ EoutT) {      // [D_OUT, E*CAP]
  constexpr int CAP = 2048, DIN = 4096, DOUT = 1024, MROWS = 16384;
  constexpr int NBN = DOUT / BN;             // 8
  constexpr int NBLK = (CAP / BM) * NBN;     // 128 blocks / expert
  constexpr int T = DIN / BK;                // 64 K-steps

  const int e    = blockIdx.x / NBLK;
  const int bid  = blockIdx.x % NBLK;
  const int row0 = (bid / NBN) * BM;
  const int col0 = (bid % NBN) * BN;

  const float* Xe = X + (size_t)(e * CAP + row0) * DIN;
  const float* We = W + (size_t)e * DIN * DOUT + col0;
  const float* Be = Bias + (size_t)e * DOUT + col0;

  __shared__ __attribute__((aligned(16))) __bf16 As[2][BM * BKP];
  __shared__ __attribute__((aligned(16))) __bf16 Bs[2][BN * BKP];

  const int tid  = threadIdx.x;
  const int lane = tid & 31;
  const int wave = tid >> 5;
  const int wm   = wave >> 2;
  const int wn   = wave & 3;
  const int half = lane >> 4;
  const int l16  = lane & 15;

  v8f acc[MT][NT];
#pragma unroll
  for (int mt = 0; mt < MT; ++mt)
#pragma unroll
    for (int nt = 0; nt < NT; ++nt)
      acc[mt][nt] = (v8f){0.f, 0.f, 0.f, 0.f, 0.f, 0.f, 0.f, 0.f};

  v4f ra[8];  // A tile 128x64 fp32: 2048 float4, 8/thread
  v4f rb[8];  // B tile 64x128 fp32: 2 4x4 micro-tiles/thread (transpose in regs)

  auto loadA = [&](int kt) {
#pragma unroll
    for (int i = 0; i < 8; ++i) {
      const int idx = tid + i * 256;
      const int r = idx >> 4, c4 = idx & 15;
      ra[i] = *(const v4f*)(Xe + (size_t)r * DIN + kt + c4 * 4);
    }
  };
  auto storeA = [&](__bf16* dstA) {
#pragma unroll
    for (int i = 0; i < 8; ++i) {
      const int idx = tid + i * 256;
      const int r = idx >> 4, c4 = idx & 15;
      v4bf t;
      t[0] = (__bf16)ra[i].x; t[1] = (__bf16)ra[i].y;
      t[2] = (__bf16)ra[i].z; t[3] = (__bf16)ra[i].w;
      *(v4bf*)(dstA + r * BKP + c4 * 4) = t;
    }
  };
  auto loadB = [&](int kt) {
#pragma unroll
    for (int i = 0; i < 2; ++i) {
      const int idx = tid + i * 256;
      const int k0 = (idx >> 5) * 4, n0 = (idx & 31) * 4;
#pragma unroll
      for (int j = 0; j < 4; ++j)
        rb[i * 4 + j] =
            *(const v4f*)(We + (size_t)(kt + k0 + j) * DOUT + n0);
    }
  };
  auto storeB = [&](__bf16* dstB) {
#pragma unroll
    for (int i = 0; i < 2; ++i) {
      const int idx = tid + i * 256;
      const int k0 = (idx >> 5) * 4, n0 = (idx & 31) * 4;
#pragma unroll
      for (int c = 0; c < 4; ++c) {  // 4x4 register transpose -> [n][k]
        v4bf t;
        t[0] = (__bf16)rb[i * 4 + 0][c]; t[1] = (__bf16)rb[i * 4 + 1][c];
        t[2] = (__bf16)rb[i * 4 + 2][c]; t[3] = (__bf16)rb[i * 4 + 3][c];
        *(v4bf*)(dstB + (n0 + c) * BKP + k0) = t;
      }
    }
  };

  // --- prologue: stage tile 0 ---
  loadA(0);
  loadB(0);
  storeA(As[0]);
  storeB(Bs[0]);
  __syncthreads();

  for (int i = 0; i < T; ++i) {
    const int cur = i & 1;
    const bool more = (i + 1) < T;
    if (more) {
      loadA((i + 1) * BK);
      loadB((i + 1) * BK);
    }
    if ((i + 2) < T)  // L2 prefetch two tiles ahead (one cacheline/thread)
      __builtin_prefetch(Xe + (size_t)(tid >> 1) * DIN + (i + 2) * BK +
                             (tid & 1) * 32, 0, 1);
    mma_slab(As[cur], Bs[cur], acc, wm, wn, half, l16);
    __syncthreads();
    if (more) {
      storeA(As[1 - cur]);
      storeB(Bs[1 - cur]);
    }
    __syncthreads();
  }

  // --- epilogue: bias + bf16, transposed store (contiguous along M) ---
#pragma unroll
  for (int nt = 0; nt < NT; ++nt) {
    const int ncol = wn * WN + nt * 16 + l16;
    const float b = Be[ncol];
    const size_t nbase = (size_t)(col0 + ncol) * MROWS + (size_t)e * CAP;
#pragma unroll
    for (int mt = 0; mt < MT; ++mt) {
      const int mrow = row0 + wm * WM + mt * 16 + half * 8;
      v8bf t;
#pragma unroll
      for (int r = 0; r < 8; ++r) t[r] = (__bf16)(acc[mt][nt][r] + b);
      *(v8bf*)(EoutT + nbase + mrow) = t;  // 16B-aligned b128 store
    }
  }
}

// ---------------------------------------------------------------------------
// Kernel 2: Out[S,N] = CW[S,K] @ EoutT^T,  K = 16384, N = 1024
// ---------------------------------------------------------------------------
__global__ __launch_bounds__(256) void combine_gemm_kernel(
    const float* __restrict__ CW,      // [S, K] fp32
    const __bf16* __restrict__ EoutT,  // [N, K] bf16
    float* __restrict__ Out) {         // [S, N] fp32
  constexpr int K = 16384, N = 1024;
  constexpr int NBN = N / BN;          // 8
  constexpr int T = K / BK;            // 256 K-steps

  const int row0 = (blockIdx.x / NBN) * BM;
  const int col0 = (blockIdx.x % NBN) * BN;

  __shared__ __attribute__((aligned(16))) __bf16 As[2][BM * BKP];
  __shared__ __attribute__((aligned(16))) __bf16 Bs[2][BN * BKP];

  const int tid  = threadIdx.x;
  const int lane = tid & 31;
  const int wave = tid >> 5;
  const int wm   = wave >> 2;
  const int wn   = wave & 3;
  const int half = lane >> 4;
  const int l16  = lane & 15;

  v8f acc[MT][NT];
#pragma unroll
  for (int mt = 0; mt < MT; ++mt)
#pragma unroll
    for (int nt = 0; nt < NT; ++nt)
      acc[mt][nt] = (v8f){0.f, 0.f, 0.f, 0.f, 0.f, 0.f, 0.f, 0.f};

  v4f ra[8];
#if !defined(HAVE_ASYNC_LDS)
  v8bf rbv[4];
#endif

  auto loadA = [&](int kt) {
#pragma unroll
    for (int i = 0; i < 8; ++i) {
      const int idx = tid + i * 256;
      const int r = idx >> 4, c4 = idx & 15;
      ra[i] = *(const v4f*)(CW + (size_t)(row0 + r) * K + kt + c4 * 4);
    }
  };
  auto storeA = [&](__bf16* dstA) {
#pragma unroll
    for (int i = 0; i < 8; ++i) {
      const int idx = tid + i * 256;
      const int r = idx >> 4, c4 = idx & 15;
      v4bf t;
      t[0] = (__bf16)ra[i].x; t[1] = (__bf16)ra[i].y;
      t[2] = (__bf16)ra[i].z; t[3] = (__bf16)ra[i].w;
      *(v4bf*)(dstA + r * BKP + c4 * 4) = t;
    }
  };
  // B tile is already bf16 and K-contiguous in EoutT: [n][k] copy, 16B units.
  // 128 rows x 128B = 1024 b128 units, 4/thread.
#if defined(HAVE_ASYNC_LDS)
  auto asyncB = [&](int kt, __bf16* dstB) {
#pragma unroll
    for (int i = 0; i < 4; ++i) {
      const int u = tid + i * 256;
      const int n = u >> 3, kg = (u & 7) * 8;
      async_copy_b128(EoutT + (size_t)(col0 + n) * K + kt + kg,
                      dstB + n * BKP + kg);
    }
  };
#else
  auto loadB = [&](int kt) {
#pragma unroll
    for (int i = 0; i < 4; ++i) {
      const int u = tid + i * 256;
      const int n = u >> 3, kg = (u & 7) * 8;
      rbv[i] = *(const v8bf*)(EoutT + (size_t)(col0 + n) * K + kt + kg);
    }
  };
  auto storeB = [&](__bf16* dstB) {
#pragma unroll
    for (int i = 0; i < 4; ++i) {
      const int u = tid + i * 256;
      const int n = u >> 3, kg = (u & 7) * 8;
      *(v8bf*)(dstB + n * BKP + kg) = rbv[i];
    }
  };
#endif

  // --- prologue ---
  loadA(0);
#if defined(HAVE_ASYNC_LDS)
  asyncB(0, Bs[0]);
#else
  loadB(0);
#endif
  storeA(As[0]);
#if defined(HAVE_ASYNC_LDS)
  wait_async_all();
#else
  storeB(Bs[0]);
#endif
  __syncthreads();

  for (int i = 0; i < T; ++i) {
    const int cur = i & 1;
    const bool more = (i + 1) < T;
    if (more) {
      loadA((i + 1) * BK);
#if defined(HAVE_ASYNC_LDS)
      asyncB((i + 1) * BK, Bs[1 - cur]);  // safe: last readers two barriers ago
#else
      loadB((i + 1) * BK);
#endif
    }
    if ((i + 2) < T)
      __builtin_prefetch(CW + (size_t)(row0 + (tid >> 1)) * K + (i + 2) * BK +
                             (tid & 1) * 32, 0, 1);
    mma_slab(As[cur], Bs[cur], acc, wm, wn, half, l16);
    __syncthreads();
    if (more) {
      storeA(As[1 - cur]);
#if defined(HAVE_ASYNC_LDS)
      wait_async_all();
#else
      storeB(Bs[1 - cur]);
#endif
    }
    __syncthreads();
  }

  // --- epilogue: fp32 store ---
#pragma unroll
  for (int nt = 0; nt < NT; ++nt) {
    const int n = col0 + wn * WN + nt * 16 + l16;
#pragma unroll
    for (int mt = 0; mt < MT; ++mt) {
      const int mrow = row0 + wm * WM + mt * 16 + half * 8;
#pragma unroll
      for (int r = 0; r < 8; ++r)
        Out[(size_t)(mrow + r) * N + n] = acc[mt][nt][r];
    }
  }
}

// ---------------------------------------------------------------------------
extern "C" void kernel_launch(void* const* d_in, const int* in_sizes, int n_in,
                              void* d_out, int out_size, void* d_ws,
                              size_t ws_size, hipStream_t stream) {
  const float* X    = (const float*)d_in[0];  // inputs          [E*CAP, D_IN]
  const float* CW   = (const float*)d_in[1];  // combine_weights [S, E*CAP]
  const float* W    = (const float*)d_in[2];  // weight          [E, D_IN, D_OUT]
  const float* Bias = (const float*)d_in[3];  // bias            [E, 1, D_OUT]
  float* Out = (float*)d_out;                 // [S, D_OUT] fp32
  __bf16* EoutT = (__bf16*)d_ws;              // [D_OUT, E*CAP] bf16 (32 MiB)

  expert_gemm_bias_kernel<<<dim3(1024), dim3(256), 0, stream>>>(X, W, Bias,
                                                                EoutT);
  combine_gemm_kernel<<<dim3(512), dim3(256), 0, stream>>>(CW, EoutT, Out);
}
